// SparseAttention_1194000908680
// MI455X (gfx1250) — matrix operand order repaired and checked by reading, not compiled, not run on hardware
//
#include <hip/hip_runtime.h>
#include <stdint.h>

typedef __attribute__((ext_vector_type(16))) _Float16 v16h;
typedef __attribute__((ext_vector_type(2)))  __fp16   v2fp16;
typedef __attribute__((ext_vector_type(8)))  float    v8f;

#define B_ 2
#define L_ 2048
#define S_ 2048
#define H_ 8
#define E_ 64
#define D_ 64

#define QROWS 64    // Q rows per workgroup
#define TS    128   // keys per S tile
#define NW    4     // waves per block (128 threads, wave32)

// single v_cvt_pk_rtz_f16_f32
__device__ __forceinline__ uint32_t pk(float a, float b) {
  union { v2fp16 h; uint32_t u; } u;
  u.h = __builtin_amdgcn_cvt_pkrtz(a, b);
  return u.u;
}

// Build a 16-halfword fragment from two 16B LDS chunks (two ds_load_b128).
__device__ __forceinline__ v16h frag8(const uint32_t* p, int o0, int o1) {
  union { uint4 q[2]; v16h h; } u;
  u.q[0] = *(const uint4*)(p + o0);
  u.q[1] = *(const uint4*)(p + o1);
  return u.h;
}

__global__ __launch_bounds__(128)
void fa_fwd_kernel(const float* __restrict__ Q, const float* __restrict__ K,
                   const float* __restrict__ V, float* __restrict__ O) {
  // LDS: K tile [s][e/2] 16KB, V tile transposed [d][s/2] 16KB,
  // Q block [m][e/2] 8KB, per-wave P scratch [16][k/2] 16KB  => 56KB of 320KB.
  __shared__ __align__(16) uint32_t ldsK[TS * (E_ / 2)];        // [128][32]
  __shared__ __align__(16) uint32_t ldsV[D_ * (TS / 2)];        // [64][64]
  __shared__ __align__(16) uint32_t ldsQ[QROWS * (E_ / 2)];     // [64][32]
  __shared__ __align__(16) uint32_t ldsP[NW * 16 * (TS / 2)];   // 4 x [16][64]

  const int tid  = threadIdx.x;
  const int wave = tid >> 5;
  const int lane = tid & 31;
  const int ln16 = lane & 15;
  const int hi   = lane >> 4;   // 0: lanes 0-15, 1: lanes 16-31

  const int nQ    = L_ / QROWS;            // 32
  const int qblk  = blockIdx.x % nQ;
  const int h     = (blockIdx.x / nQ) % H_;
  const int b     = blockIdx.x / (nQ * H_);
  const int qbase = qblk * QROWS;

  // Uniform (scalar) bases; all per-lane offsets below are 32-bit ints so the
  // backend can use SADDR + 32-bit VGPR-offset global addressing.
  const float* Qb = Q + (size_t)b * L_ * 512 + (size_t)qbase * 512 + h * 64;
  const float* Kb = K + (size_t)b * S_ * 512 + h * 64;
  const float* Vb = V + (size_t)b * S_ * 512 + h * 64;

  // ---- stage Q as packed f16, pre-scaled by (1/sqrt(E)) * log2(e) so the
  // softmax exponentials are pure exp2 ----
  {
    const float scale = 0.125f * 1.44269504088896340736f;
    #pragma unroll
    for (int i = 0; i < 8; ++i) {          // 1024 float4-groups / 128 threads
      int g  = tid + i * 128;
      int m  = g >> 4;                      // q row in block
      int dq = g & 15;                      // float4 index within row
      float4 q4 = *(const float4*)(Qb + m * 512 + dq * 4);
      uint2 pr;
      pr.x = pk(q4.x * scale, q4.y * scale);
      pr.y = pk(q4.z * scale, q4.w * scale);
      *(uint2*)(ldsQ + m * 32 + dq * 2) = pr;
    }
  }
  __syncthreads();

  // ---- per-wave Q A-fragments (16x32 f16 each), kept in registers ----
  // A frag, contiguous k order used here (B matches): lanes 0-15 hold M=lane, K=0..15; lanes 16-31 same M, K=16..31
  const int mq = wave * 16 + ln16;
  const v16h aQ0 = frag8(ldsQ, mq * 32 + 0  + hi * 8, mq * 32 + 0  + hi * 8 + 4); // e 0..31
  const v16h aQ1 = frag8(ldsQ, mq * 32 + 16 + hi * 8, mq * 32 + 16 + hi * 8 + 4); // e 32..63

  // All-ones B fragment: o4 = P x Ones gives the softmax row-sum in every lane,
  // rescaled by the same corr chain as o0..o3 (row-sum via the matrix unit).
  v16h bOnes;
  #pragma unroll
  for (int i = 0; i < 16; ++i) bOnes[i] = (_Float16)1.0f;

  v8f o0 = {}, o1 = {}, o2 = {}, o3 = {}, o4 = {};
  float runmax[8];
  #pragma unroll
  for (int r = 0; r < 8; ++r) runmax[r] = -1e30f;

  _Float16*       Pw = ((_Float16*)ldsP) + wave * 16 * TS;      // halfword view
  const uint32_t* Pd = ldsP + wave * 16 * (TS / 2);             // dword view

  for (int kb = 0; kb < S_; kb += TS) {
    const float* Kt = Kb + (size_t)kb * 512;   // uniform per-tile base
    const float* Vt = Vb + (size_t)kb * 512;

    __syncthreads();  // previous tile fully consumed

    // ---- stage K tile: ldsK[s*32 + e/2], float4 global loads ----
    #pragma unroll
    for (int i = 0; i < 16; ++i) {         // 2048 float4-groups
      int g   = tid + i * 128;
      int s   = g >> 4;
      int dq  = g & 15;
      int off = s * 512 + dq * 4;          // loop-invariant lane offset
      float4 k4 = *(const float4*)(Kt + off);
      uint2 pr;
      pr.x = pk(k4.x, k4.y);
      pr.y = pk(k4.z, k4.w);
      *(uint2*)(ldsK + s * 32 + dq * 2) = pr;
      if (kb + TS < S_) __builtin_prefetch(Kt + off + TS * 512, 0, 1);
    }
    // ---- stage V tile transposed: ldsV[d*64 + s/2] (f16 pairs along s) ----
    #pragma unroll
    for (int i = 0; i < 8; ++i) {          // 1024 groups: (4 d-values) x (s pair)
      int g   = tid + i * 128;
      int d4  = g >> 6;                     // d block of 4 (0..15)
      int sp  = g & 63;                     // s pair index (0..63)
      int off = sp * 1024 + d4 * 4;         // s = 2*sp rows of 512 floats
      float4 va = *(const float4*)(Vt + off);
      float4 vb = *(const float4*)(Vt + off + 512);
      ldsV[(d4 * 4 + 0) * 64 + sp] = pk(va.x, vb.x);
      ldsV[(d4 * 4 + 1) * 64 + sp] = pk(va.y, vb.y);
      ldsV[(d4 * 4 + 2) * 64 + sp] = pk(va.z, vb.z);
      ldsV[(d4 * 4 + 3) * 64 + sp] = pk(va.w, vb.w);
      if (kb + TS < S_) __builtin_prefetch(Vt + off + TS * 512, 0, 1);
    }
    __syncthreads();  // tile visible

    // ---- scores: S = Q * K^T  (8 N-groups of 16 keys, 2 chained K-steps) ----
    v8f c[8];
    #pragma unroll
    for (int j = 0; j < 8; ++j) {
      int s = j * 16 + ln16;  // B frag, k order matching aQ: lane column = key; lanes 0-15 e 0..15, 16-31 e 16..31
      v16h bK0 = frag8(ldsK, s * 32 + hi * 8,      s * 32 + hi * 8 + 4);       // e 0..31
      v16h bK1 = frag8(ldsK, s * 32 + 16 + hi * 8, s * 32 + 16 + hi * 8 + 4);  // e 32..63
      v8f z = {};
      z = __builtin_amdgcn_wmma_f32_16x16x32_f16(false, aQ0, false, bK0, (short)0, z, false, false);
      z = __builtin_amdgcn_wmma_f32_16x16x32_f16(false, aQ1, false, bK1, (short)0, z, false, false);
      c[j] = z;
    }

    // ---- online softmax, exp2 domain (C layout: vgpr r -> row r+8*hi, lane%16 -> col) ----
    #pragma unroll
    for (int r = 0; r < 8; ++r) {
      float tm = fmaxf(fmaxf(fmaxf(c[0][r], c[1][r]), fmaxf(c[2][r], c[3][r])),
                       fmaxf(fmaxf(c[4][r], c[5][r]), fmaxf(c[6][r], c[7][r])));
      tm = fmaxf(tm, __shfl_xor(tm, 1, 32));
      tm = fmaxf(tm, __shfl_xor(tm, 2, 32));
      tm = fmaxf(tm, __shfl_xor(tm, 4, 32));
      tm = fmaxf(tm, __shfl_xor(tm, 8, 32));
      float nm   = fmaxf(runmax[r], tm);
      float corr = __builtin_amdgcn_exp2f(runmax[r] - nm);
      runmax[r]  = nm;

      o0[r] *= corr; o1[r] *= corr; o2[r] *= corr; o3[r] *= corr; o4[r] *= corr;

      int row = r + hi * 8;
      #pragma unroll
      for (int j = 0; j < 8; ++j) {
        float p = __builtin_amdgcn_exp2f(c[j][r] - nm);
        Pw[row * TS + j * 16 + ln16] = (_Float16)p;
      }
    }
    __syncthreads();  // P visible for A-layout re-read

    // ---- O += P * V, rowsum += P * 1  (4 K-steps of 32 keys) ----
    #pragma unroll
    for (int ks = 0; ks < 4; ++ks) {
      v16h aP = frag8(Pd, ln16 * 64 + ks * 16 + hi * 8,
                          ln16 * 64 + ks * 16 + hi * 8 + 4);
      {
        int d = 0 * 16 + ln16;
        v16h bV = frag8(ldsV, d * 64 + ks * 16 + hi * 8, d * 64 + ks * 16 + hi * 8 + 4);
        o0 = __builtin_amdgcn_wmma_f32_16x16x32_f16(false, aP, false, bV, (short)0, o0, false, false);
      }
      {
        int d = 1 * 16 + ln16;
        v16h bV = frag8(ldsV, d * 64 + ks * 16 + hi * 8, d * 64 + ks * 16 + hi * 8 + 4);
        o1 = __builtin_amdgcn_wmma_f32_16x16x32_f16(false, aP, false, bV, (short)0, o1, false, false);
      }
      {
        int d = 2 * 16 + ln16;
        v16h bV = frag8(ldsV, d * 64 + ks * 16 + hi * 8, d * 64 + ks * 16 + hi * 8 + 4);
        o2 = __builtin_amdgcn_wmma_f32_16x16x32_f16(false, aP, false, bV, (short)0, o2, false, false);
      }
      {
        int d = 3 * 16 + ln16;
        v16h bV = frag8(ldsV, d * 64 + ks * 16 + hi * 8, d * 64 + ks * 16 + hi * 8 + 4);
        o3 = __builtin_amdgcn_wmma_f32_16x16x32_f16(false, aP, false, bV, (short)0, o3, false, false);
      }
      o4 = __builtin_amdgcn_wmma_f32_16x16x32_f16(false, aP, false, bOnes, (short)0, o4, false, false);
    }
  }

  // ---- epilogue: normalize by o4 (rowsum, identical in all 16 lanes of the half)
  // and scatter fp32 to [B, L, H, D] ----
  #pragma unroll
  for (int r = 0; r < 8; ++r) {
    float inv = 1.0f / o4[r];
    int m = qbase + wave * 16 + r + hi * 8;
    float* og = O + ((size_t)(b * L_ + m) * H_ + h) * D_;
    og[ 0 + ln16] = o0[r] * inv;
    og[16 + ln16] = o1[r] * inv;
    og[32 + ln16] = o2[r] * inv;
    og[48 + ln16] = o3[r] * inv;
  }
}

extern "C" void kernel_launch(void* const* d_in, const int* in_sizes, int n_in,
                              void* d_out, int out_size, void* d_ws, size_t ws_size,
                              hipStream_t stream) {
  (void)in_sizes; (void)n_in; (void)out_size; (void)d_ws; (void)ws_size;
  const float* Q = (const float*)d_in[0];
  const float* K = (const float*)d_in[1];
  const float* V = (const float*)d_in[2];
  float* O = (float*)d_out;

  dim3 grid(B_ * H_ * (L_ / QROWS));  // 512 workgroups
  dim3 block(128);                    // 4 wave32 waves
  fa_fwd_kernel<<<grid, block, 0, stream>>>(Q, K, V, O);
}
